// GatedDIP_12824772346337
// MI455X (gfx1250) — compile-verified, matrix operand order checked by reading, not run on hardware
//
#include <hip/hip_runtime.h>
#include <hip/hip_bf16.h>
#include <math.h>

// ---------------------------------------------------------------- constants
#define BB 8
#define CC 3
#define HH 512
#define WW 512
#define HW (HH*WW)            // 262144
#define NPLANE (BB*CC)        // 24
#define NTOT (NPLANE*HW)      // 6291456
#define KS 13
#define KR 6
#define HB 2048               // dark-channel histogram bins
#define NUMPX 262             // max(HW//1000,1)
#define PI_F 3.14159265358979f

// workspace layout (float offsets)
#define OFF_GATE   0          // [8][8]
#define OFF_WB     64         // [8][3]
#define OFF_GAMMA  88         // [8]
#define OFF_SHARP  96         // [8]
#define OFF_OMEGA  104        // [8]
#define OFF_ALPHA  112        // [8]
#define OFF_TC     120        // [8][8]
#define OFF_TINV   184        // [8]
#define OFF_LINE   192        // [8] lineShift[b] = line[(b+4)%8]
#define OFF_T      200        // [8] dark threshold
#define OFF_ASUM   208        // [8][3] atmospheric light sums
#define OFF_MM     232        // 14 uint minmax keys: pairs for comps 0..6
#define OFF_CCNT   248        // [8] int candidate counts
#define OFF_HIST   512        // [8][HB] int
#define OFF_CAND   (512 + BB*HB)        // [8][1024] float
#define CTRL_FLOATS 32768
#define OFF_DARK   CTRL_FLOATS                   // [8][HW]
#define OFF_BLUR   (OFF_DARK + BB*HW)            // [24][HW]
#define OFF_CBUF   (OFF_BLUR + NPLANE*HW)        // [24][HW] float2

typedef __attribute__((ext_vector_type(16))) _Float16 v16h;
typedef __attribute__((ext_vector_type(8)))  float    v8f;
typedef __attribute__((ext_vector_type(2)))  float    v2f;

#if defined(__has_builtin)
#  if __has_builtin(__builtin_amdgcn_wmma_f32_16x16x4_f32)
#    define HAVE_WMMA_F32 1
#  endif
#endif

// ---------------------------------------------------------------- helpers
__device__ __forceinline__ unsigned fkey(float f) {
  unsigned u = __float_as_uint(f);
  return (u & 0x80000000u) ? ~u : (u | 0x80000000u);
}
__device__ __forceinline__ float funkey(unsigned k) {
  unsigned u = (k & 0x80000000u) ? (k & 0x7FFFFFFFu) : ~k;
  return __uint_as_float(u);
}
__device__ __forceinline__ int reflect512(int i) {
  if (i < 0) i = -i;
  if (i >= 512) i = 1022 - i;
  return i;
}
__device__ __forceinline__ int brev9(int i) { return (int)(__brev((unsigned)i) >> 23); }

__device__ __forceinline__ float tanh01f(float v) { return tanhf(v) * 0.5f + 0.5f; }

// concatenated weight column mapping: 0-7 gate, 8-10 wb, 11 gamma, 12 sharp,
// 13 defog, 14 contrast, 15-22 tone, 23 fft.
// Per-lane one-time column select: W[k][n] == ptr[k*stride]
__device__ __forceinline__ const float* colSel(int n, int* stride,
    const float* gw, const float* wbw, const float* gaw, const float* shw,
    const float* dfw, const float* ctw, const float* tow, const float* ffw) {
  if (n < 8)   { *stride = 8; return gw + n; }
  if (n < 11)  { *stride = 3; return wbw + (n - 8); }
  if (n == 11) { *stride = 1; return gaw; }
  if (n == 12) { *stride = 1; return shw; }
  if (n == 13) { *stride = 1; return dfw; }
  if (n == 14) { *stride = 1; return ctw; }
  if (n < 23)  { *stride = 8; return tow + (n - 15); }
  *stride = 1; return ffw;   // n == 23
}
__device__ float getB(int n,
                      const float* gb, const float* wbb, const float* gab,
                      const float* shb, const float* dfb, const float* ctb,
                      const float* tob, const float* ffb) {
  if (n < 8)   return gb[n];
  if (n < 11)  return wbb[n-8];
  if (n == 11) return gab[0];
  if (n == 12) return shb[0];
  if (n == 13) return dfb[0];
  if (n == 14) return ctb[0];
  if (n < 23)  return tob[n-15];
  if (n == 23) return ffb[0];
  return 0.f;
}

// ISA 16-bit A/B fragment K mapping (16x32): lane<16 -> K chunks {0..7,16..23},
// lane>=16 -> {8..15,24..31}; half h -> VGPR h/2, K pair element h&1.
__device__ __forceinline__ int khalf(int h) {
  int vg = h >> 1, p = h & 1;
  return (vg < 4) ? (2*vg + p) : (16 + 2*(vg-4) + p);
}

// ---------------------------------------------------------------- kernel 0: init
__global__ void k_init(float* ws) {
  int tid = threadIdx.x;
  int* hist = (int*)(ws + OFF_HIST);
  for (int i = tid; i < BB*HB; i += 256) hist[i] = 0;
  int* ccnt = (int*)(ws + OFF_CCNT);
  if (tid < BB) ccnt[tid] = 0;
  if (tid < BB*CC) ws[OFF_ASUM + tid] = 0.f;
  unsigned* mm = (unsigned*)(ws + OFF_MM);
  if (tid < 7) { mm[2*tid] = 0xFFFFFFFFu; mm[2*tid + 1] = 0u; }
}

// ---------------------------------------------------------------- kernel 1: params via WMMA
// One wave computes z[8][24] = latent @ Wcat + bcat, then derives all params.
// Preferred: exact f32 path with v_wmma_f32_16x16x4_f32; fallback f16 (confirmed).
__global__ void __launch_bounds__(32)
k_params(const float* latent,
         const float* gw, const float* gb, const float* wbw, const float* wbb,
         const float* gaw, const float* gab, const float* shw, const float* shb,
         const float* dfw, const float* dfb, const float* ctw, const float* ctb,
         const float* tow, const float* tob, const float* ffw, const float* ffb,
         float* ws) {
  int lane = threadIdx.x;
  int m = lane & 15;
  v8f c0 = {}; v8f c1 = {};

  int s0, s1;
  const float* p0 = colSel(m, &s0, gw, wbw, gaw, shw, dfw, ctw, tow, ffw);
  int n1 = 16 + m;
  bool v1ok = (n1 < 24);
  const float* p1 = colSel(v1ok ? n1 : 23, &s1, gw, wbw, gaw, shw, dfw, ctw, tow, ffw);
  const float* arow = latent + m * 256;
  bool aok = (m < 8);

#ifdef HAVE_WMMA_F32
  // A 16x4 f32: lane<16 holds K=0,1; lane>=16 holds K=2,3 (B mirrored on N=lane&15)
  int kb = (lane & 16) ? 2 : 0;
  for (int k0 = 0; k0 < 256; k0 += 4) {
    int k = k0 + kb;
    v2f af, bf0, bf1;
    af[0]  = aok ? arow[k]     : 0.f;
    af[1]  = aok ? arow[k + 1] : 0.f;
    bf0[0] = p0[k * s0];
    bf0[1] = p0[(k + 1) * s0];
    bf1[0] = v1ok ? p1[k * s1]       : 0.f;
    bf1[1] = v1ok ? p1[(k + 1) * s1] : 0.f;
    c0 = __builtin_amdgcn_wmma_f32_16x16x4_f32(false, af, false, bf0, (short)0, c0, false, false);
    c1 = __builtin_amdgcn_wmma_f32_16x16x4_f32(false, af, false, bf1, (short)0, c1, false, false);
  }
#else
  int kb = (lane & 16) ? 8 : 0;
  for (int k0 = 0; k0 < 256; k0 += 32) {
    v16h a; v16h b0; v16h b1;
#pragma unroll
    for (int h = 0; h < 16; ++h) {
      int k = k0 + kb + khalf(h);
      a[h]  = aok ? (_Float16)arow[k] : (_Float16)0.f;
      b0[h] = (_Float16)p0[k * s0];
      b1[h] = v1ok ? (_Float16)p1[k * s1] : (_Float16)0.f;
    }
    c0 = __builtin_amdgcn_wmma_f32_16x16x32_f16(false, a, false, b0, (short)0, c0, false, false);
    c1 = __builtin_amdgcn_wmma_f32_16x16x32_f16(false, a, false, b1, (short)0, c1, false, false);
  }
#endif

  __shared__ float z[8][24];
  __shared__ float lineRaw[8];
  if (lane < 16) {
#pragma unroll
    for (int r = 0; r < 8; ++r) {
      z[r][lane] = c0[r] + getB(lane, gb, wbb, gab, shb, dfb, ctb, tob, ffb);
      if (lane < 8)
        z[r][16 + lane] = c1[r] + getB(16 + lane, gb, wbb, gab, shb, dfb, ctb, tob, ffb);
    }
  }
  __syncthreads();
  if (lane < 8) {
    int b = lane;
    for (int i = 0; i < 8; ++i)
      ws[OFF_GATE + b*8 + i] = tanh01f(z[b][i]) * 0.99f + 0.01f;
    float wb0 = __expf(tanh01f(z[b][8])  * 1.0f - 0.5f);
    float wb1 = __expf(tanh01f(z[b][9])  * 1.0f - 0.5f);
    float wb2 = __expf(tanh01f(z[b][10]) * 1.0f - 0.5f);
    float cs = 1.0f / (1e-5f + 0.27f*wb0 + 0.67f*wb1 + 0.06f*wb2);
    ws[OFF_WB + b*3 + 0] = cs*wb0;
    ws[OFF_WB + b*3 + 1] = cs*wb1;
    ws[OFF_WB + b*3 + 2] = cs*wb2;
    const float lg = 0.9162907318741551f;  // log(2.5)
    ws[OFF_GAMMA + b] = __expf(tanh01f(z[b][11]) * (2.0f*lg) - lg);
    ws[OFF_SHARP + b] = tanh01f(z[b][12]) * 0.9f + 0.1f;
    ws[OFF_OMEGA + b] = tanh01f(z[b][13]) * 0.9f + 0.1f;
    ws[OFF_ALPHA + b] = tanhf(z[b][14]);
    float tsum = 1e-30f;
    for (int i = 0; i < 8; ++i) {
      float t = tanh01f(z[b][15 + i]) * 1.5f + 0.5f;
      ws[OFF_TC + b*8 + i] = t;
      tsum += t;
    }
    ws[OFF_TINV + b] = 8.0f / tsum;
    float rate = tanh01f(z[b][23]) * 0.7f + 0.1f;
    lineRaw[b] = floorf(sqrtf((float)HW * rate) * 0.5f);
  }
  __syncthreads();
  // torch fftshift over ALL axes: batch b's spectrum gets mask of line[(b+4)%8]
  if (lane < 8) ws[OFF_LINE + lane] = lineRaw[(lane + 4) & 7];
}

// ---------------------------------------------------------------- kernel 2: tiled separable blur (reflect)
__global__ void __launch_bounds__(256)
k_blur(const float* x, float* ws) {
  __shared__ float tin[44][44];
  __shared__ float hb[44][32];
  int tid = threadIdx.x;
  int plane = blockIdx.z;
  int ox = blockIdx.x * 32 - KR;
  int oy = blockIdx.y * 32 - KR;
  const float* src = x + (size_t)plane * HW;
  float* blur = ws + OFF_BLUR + (size_t)plane * HW;

  float wgt[KS]; float s = 0.f;
  for (int i = 0; i < KS; ++i) { wgt[i] = __expf(-(float)((i-KR)*(i-KR)) / 8.0f); s += wgt[i]; }
  for (int i = 0; i < KS; ++i) wgt[i] /= s;

  for (int li = tid; li < 44*44; li += 256) {
    int r = li / 44, cx = li % 44;
    int gy = reflect512(oy + r), gx = reflect512(ox + cx);
    tin[r][cx] = src[gy*WW + gx];
  }
  __syncthreads();
  for (int li = tid; li < 44*32; li += 256) {
    int r = li / 32, cx = li % 32;
    float acc = 0.f;
    for (int t = 0; t < KS; ++t) acc += wgt[t] * tin[r][cx + t];
    hb[r][cx] = acc;
  }
  __syncthreads();
  for (int li = tid; li < 32*32; li += 256) {
    int yy = li / 32, xx = li % 32;
    float acc = 0.f;
    for (int t = 0; t < KS; ++t) acc += wgt[t] * hb[yy + t][xx];
    blur[(blockIdx.y*32 + yy)*WW + blockIdx.x*32 + xx] = acc;
  }
}

// ---------------------------------------------------------------- kernel 3: dark channel + histogram
__global__ void __launch_bounds__(256)
k_dark(const float* x, float* ws) {
  int b = blockIdx.y;
  int pix = blockIdx.x * 256 + threadIdx.x;
  const float* xb = x + (size_t)b * 3 * HW + pix;
  float d = fminf(xb[0], fminf(xb[HW], xb[2*HW]));
  ws[OFF_DARK + (size_t)b*HW + pix] = d;
  int bin = (int)(d * HB); bin = bin < 0 ? 0 : (bin > HB-1 ? HB-1 : bin);
  atomicAdd((int*)(ws + OFF_HIST) + b*HB + bin, 1);
}

// ---------------------------------------------------------------- kernel 4: exact 262nd-largest threshold
__global__ void __launch_bounds__(256)
k_select(float* ws) {
  int b = blockIdx.x, tid = threadIdx.x;
  __shared__ int sAbove, sBin;
  const int* hist = (const int*)(ws + OFF_HIST) + b*HB;
  if (tid == 0) {
    int cum = 0, bin = HB - 1, above = 0;
    for (; bin >= 0; --bin) {
      int c = hist[bin];
      if (cum + c >= NUMPX) { above = cum; break; }
      cum += c;
    }
    if (bin < 0) { bin = 0; above = cum; }
    sAbove = above; sBin = bin;
  }
  __syncthreads();
  float* cand = ws + OFF_CAND + b*1024;
  int* ccnt = (int*)(ws + OFF_CCNT) + b;
  const float* dark = ws + OFF_DARK + (size_t)b*HW;
  for (int pix = tid; pix < HW; pix += 256) {
    float d = dark[pix];
    int bin = (int)(d * HB); bin = bin < 0 ? 0 : (bin > HB-1 ? HB-1 : bin);
    if (bin == sBin) {
      int i = atomicAdd(ccnt, 1);
      if (i < 1024) cand[i] = d;
    }
  }
  __threadfence_block();
  __syncthreads();
  if (tid == 0) ws[OFF_T + b] = (float)sBin / (float)HB;  // fallback
  __threadfence_block();
  __syncthreads();
  int n = *ccnt; if (n > 1024) n = 1024;
  int need = NUMPX - sAbove;  // kth-largest within bin, k = need
  for (int i = tid; i < n; i += 256) {
    float v = cand[i];
    int gt = 0;
    for (int j = 0; j < n; ++j) {
      float u = cand[j];
      gt += (u > v) || (u == v && j < i);
    }
    if (gt == need - 1) ws[OFF_T + b] = v;  // exact threshold (262nd-largest)
  }
}

// ---------------------------------------------------------------- kernel 5: atmospheric light sums
__global__ void __launch_bounds__(256)
k_atmo(const float* x, float* ws) {
  __shared__ float sr[3][256];
  int b = blockIdx.y, tid = threadIdx.x;
  int pix = blockIdx.x * 256 + tid;
  float T = ws[OFF_T + b];
  float d = ws[OFF_DARK + (size_t)b*HW + pix];
  const float* xb = x + (size_t)b * 3 * HW + pix;
  bool take = d > T;
  sr[0][tid] = take ? xb[0]      : 0.f;
  sr[1][tid] = take ? xb[HW]     : 0.f;
  sr[2][tid] = take ? xb[2*HW]   : 0.f;
  __syncthreads();
  for (int s = 128; s > 0; s >>= 1) {
    if (tid < s) { sr[0][tid] += sr[0][tid+s]; sr[1][tid] += sr[1][tid+s]; sr[2][tid] += sr[2][tid+s]; }
    __syncthreads();
  }
  if (tid < 3) atomicAdd(ws + OFF_ASUM + b*3 + tid, sr[tid][0]);
}

// ---------------------------------------------------------------- FFT core (512-pt radix-2 DIT, bit-reversed input)
__device__ void fft512(float2* buf, int tid, float sign) {
  for (int s = 1; s <= 9; ++s) {
    int mh = 1 << (s - 1);
    int k = tid & (mh - 1);
    int grp = tid >> (s - 1);
    int i0 = (grp << s) + k;
    int i1 = i0 + mh;
    float ang = sign * 6.283185307179586f * (float)k / (float)(1 << s);
    float sw, cw; __sincosf(ang, &sw, &cw);
    float2 u = buf[i0], v = buf[i1];
    float2 t = { cw*v.x - sw*v.y, cw*v.y + sw*v.x };
    buf[i0] = make_float2(u.x + t.x, u.y + t.y);
    buf[i1] = make_float2(u.x - t.x, u.y - t.y);
    __syncthreads();
  }
}

// kernel 6: forward FFT along rows. Input row streamed into LDS (bit-reversed
// scatter, re slots) via CDNA5 async global->LDS loads (ASYNCcnt tracked);
// forward 1/512 scale folded into the (linear) output store.
__global__ void __launch_bounds__(256)
k_fft_rows(const float* x, float* ws) {
  __shared__ float2 buf[512];
  int tid = threadIdx.x;
  int plane = blockIdx.x >> 9;
  int y = blockIdx.x & 511;
  const float* src = x + (size_t)plane * HW + (size_t)y * WW;
  for (int i = tid; i < 512; i += 256) {
    unsigned lofs = (unsigned)(size_t)&buf[brev9(i)];       // LDS byte addr of re slot
    unsigned long long ga = (unsigned long long)(size_t)(src + i);
    asm volatile("global_load_async_to_lds_b32 %0, %1, off"
                 :: "v"(lofs), "v"(ga) : "memory");
    buf[brev9(i)].y = 0.f;
  }
  asm volatile("s_wait_asynccnt 0x0" ::: "memory");
  __syncthreads();
  fft512(buf, tid, -1.0f);
  const float sc = 1.0f / 512.0f;
  float2* cb = (float2*)(ws + OFF_CBUF) + (size_t)plane * HW + (size_t)y * WW;
  for (int i = tid; i < 512; i += 256)
    cb[i] = make_float2(buf[i].x * sc, buf[i].y * sc);
}

// kernel 7: forward col FFT (scale 1/512), high-pass mask, inverse col FFT
__global__ void __launch_bounds__(256)
k_fft_cols(float* ws) {
  __shared__ float2 buf[512];
  int tid = threadIdx.x;
  int plane = blockIdx.x >> 9;
  int kc = blockIdx.x & 511;
  int b = plane / 3;
  int line = (int)ws[OFF_LINE + b];
  float2* cb = (float2*)(ws + OFF_CBUF) + (size_t)plane * HW + kc;
  const float sc = 1.0f / 512.0f;
  for (int i = tid; i < 512; i += 256) {
    float2 v = cb[(size_t)i * WW];
    buf[brev9(i)] = make_float2(v.x * sc, v.y * sc);
  }
  __syncthreads();
  fft512(buf, tid, -1.0f);
  bool lowC = (kc < line) || (kc >= 512 - line);
  if (lowC) {
    for (int i = tid; i < 512; i += 256)
      if (i < line || i >= 512 - line) buf[i] = make_float2(0.f, 0.f);
  }
  __syncthreads();
  // re-bit-reverse in place for the inverse pass
  float2 a0 = buf[tid], a1 = buf[tid + 256];
  __syncthreads();
  buf[brev9(tid)] = a0; buf[brev9(tid + 256)] = a1;
  __syncthreads();
  fft512(buf, tid, +1.0f);   // ifft, norm='forward' -> no scale
  for (int i = tid; i < 512; i += 256) cb[(size_t)i * WW] = buf[i];
}

// ---------------------------------------------------------------- kernel 8: component min/max stats
__global__ void __launch_bounds__(256)
k_stats(const float* x, float* ws) {
  __shared__ float sred[256];
  int tid = threadIdx.x;
  int gid = blockIdx.x * 256 + tid;
  int b = gid / HW, pix = gid % HW;
  const float* xb = x + (size_t)b * 3 * HW + pix;
  const float* blb = ws + OFF_BLUR + (size_t)b * 3 * HW + pix;
  float xc[3] = { xb[0], xb[HW], xb[2*HW] };
  float bl[3] = { blb[0], blb[HW], blb[2*HW] };

  float gamma = ws[OFF_GAMMA + b], yshp = ws[OFF_SHARP + b];
  float omega = ws[OFF_OMEGA + b], alpha = ws[OFF_ALPHA + b];
  float tinv = ws[OFF_TINV + b];
  float a[3], wb[3], tc[8];
  for (int c = 0; c < 3; ++c) { a[c] = ws[OFF_ASUM + b*3 + c] / (float)NUMPX; wb[c] = ws[OFF_WB + b*3 + c]; }
  for (int i = 0; i < 8; ++i) tc[i] = ws[OFF_TC + b*8 + i];

  float lum = 0.27f*xc[0] + 0.67f*xc[1] + 0.06f*xc[2];
  lum = fminf(fmaxf(lum, 0.f), 1.f);
  float clum = -__cosf(PI_F * lum) * 0.5f + 0.5f;
  float ratio = fminf(xc[0]/a[0], fminf(xc[1]/a[1], xc[2]/a[2]));
  float tfog = fmaxf(1.0f - omega * ratio, 0.01f);

  float mn[6], mx[6];
  for (int s = 0; s < 6; ++s) { mn[s] = 3.4e38f; mx[s] = -3.4e38f; }
  for (int c = 0; c < 3; ++c) {
    float v;
    v = wb[c] * xc[c];                                   mn[0]=fminf(mn[0],v); mx[0]=fmaxf(mx[0],v);
    v = powf(fmaxf(xc[c], 1e-4f), gamma);                mn[1]=fminf(mn[1],v); mx[1]=fmaxf(mx[1],v);
    v = xc[c] + yshp * (xc[c] - bl[c]);                  mn[2]=fminf(mn[2],v); mx[2]=fmaxf(mx[2],v);
    v = (xc[c] - a[c]) / tfog + a[c];                    mn[3]=fminf(mn[3],v); mx[3]=fmaxf(mx[3],v);
    float ci = xc[c] / (lum + 1e-6f) * clum;
    v = (1.0f - alpha) * xc[c] + alpha * ci;             mn[4]=fminf(mn[4],v); mx[4]=fmaxf(mx[4],v);
    float tt = 0.f;
    for (int i = 0; i < 8; ++i)
      tt += fminf(fmaxf(xc[c] - (float)i * 0.125f, 0.f), 0.125f) * tc[i];
    v = tt * tinv;                                       mn[5]=fminf(mn[5],v); mx[5]=fmaxf(mx[5],v);
  }
  unsigned* mm = (unsigned*)(ws + OFF_MM);
  for (int s = 0; s < 6; ++s) {
    sred[tid] = mn[s]; __syncthreads();
    for (int st = 128; st > 0; st >>= 1) { if (tid < st) sred[tid] = fminf(sred[tid], sred[tid+st]); __syncthreads(); }
    if (tid == 0) atomicMin(&mm[2*s], fkey(sred[0]));
    __syncthreads();
    sred[tid] = mx[s]; __syncthreads();
    for (int st = 128; st > 0; st >>= 1) { if (tid < st) sred[tid] = fmaxf(sred[tid], sred[tid+st]); __syncthreads(); }
    if (tid == 0) atomicMax(&mm[2*s + 1], fkey(sred[0]));
    __syncthreads();
  }
}

// ---------------------------------------------------------------- kernel 9: inverse row FFT fused with gated sum
__global__ void __launch_bounds__(256)
k_sum(const float* x, float* ws, float* out) {
  __shared__ float2 buf[512];
  __shared__ float sred[256];
  int tid = threadIdx.x;
  int plane = blockIdx.x >> 9;
  int y = blockIdx.x & 511;
  int b = plane / 3, c = plane % 3;

  float2* cb = (float2*)(ws + OFF_CBUF) + (size_t)plane * HW + (size_t)y * WW;
  for (int i = tid; i < 512; i += 256) buf[brev9(i)] = cb[i];
  __syncthreads();
  fft512(buf, tid, +1.0f);   // final inverse (row) FFT

  float gate[8];
  for (int i = 0; i < 8; ++i) gate[i] = ws[OFF_GATE + b*8 + i];
  unsigned* mm = (unsigned*)(ws + OFF_MM);
  float mn[6], inv[6];
  for (int s = 0; s < 6; ++s) {
    mn[s] = funkey(mm[2*s]);
    float mx = funkey(mm[2*s + 1]);
    inv[s] = 1.0f / (mx - mn[s]);
  }
  float gamma = ws[OFF_GAMMA + b], yshp = ws[OFF_SHARP + b];
  float omega = ws[OFF_OMEGA + b], alpha = ws[OFF_ALPHA + b];
  float tinv = ws[OFF_TINV + b];
  float a[3], tc[8];
  for (int cc2 = 0; cc2 < 3; ++cc2) a[cc2] = ws[OFF_ASUM + b*3 + cc2] / (float)NUMPX;
  for (int i = 0; i < 8; ++i) tc[i] = ws[OFF_TC + b*8 + i];
  float wbc = ws[OFF_WB + b*3 + c];

  const float* xrow0 = x + (size_t)b * 3 * HW + (size_t)y * WW;
  const float* blrow = ws + OFF_BLUR + (size_t)plane * HW + (size_t)y * WW;
  __builtin_prefetch(xrow0, 0, 1);
  __builtin_prefetch(blrow, 0, 1);
  float lmn = 3.4e38f, lmx = -3.4e38f;
  for (int i = tid; i < 512; i += 256) {
    float x0 = xrow0[i], x1 = xrow0[HW + i], x2 = xrow0[2*HW + i];
    float xc = (c == 0) ? x0 : (c == 1) ? x1 : x2;
    float bl = blrow[i];
    float lum = fminf(fmaxf(0.27f*x0 + 0.67f*x1 + 0.06f*x2, 0.f), 1.f);
    float clum = -__cosf(PI_F * lum) * 0.5f + 0.5f;
    float ratio = fminf(x0/a[0], fminf(x1/a[1], x2/a[2]));
    float tfog = fmaxf(1.0f - omega * ratio, 0.01f);

    float sum = xc * gate[2];                                    // identity
    sum += (wbc * xc - mn[0]) * inv[0] * gate[0];                // white balance
    sum += (powf(fmaxf(xc, 1e-4f), gamma) - mn[1]) * inv[1] * gate[1];  // gamma
    sum += (xc + yshp * (xc - bl) - mn[2]) * inv[2] * gate[3];   // sharpen
    sum += ((xc - a[c]) / tfog + a[c] - mn[3]) * inv[3] * gate[4];  // defog
    float ci = xc / (lum + 1e-6f) * clum;
    sum += ((1.0f - alpha) * xc + alpha * ci - mn[4]) * inv[4] * gate[5];  // contrast
    float tt = 0.f;
    for (int k = 0; k < 8; ++k)
      tt += fminf(fmaxf(xc - (float)k * 0.125f, 0.f), 0.125f) * tc[k];
    sum += (tt * tinv - mn[5]) * inv[5] * gate[6];               // tone curve
    sum += fabsf(buf[i].x) * gate[7];                            // fft high-pass
    out[(size_t)plane * HW + (size_t)y * WW + i] = sum;
    lmn = fminf(lmn, sum); lmx = fmaxf(lmx, sum);
  }
  sred[tid] = lmn; __syncthreads();
  for (int st = 128; st > 0; st >>= 1) { if (tid < st) sred[tid] = fminf(sred[tid], sred[tid+st]); __syncthreads(); }
  if (tid == 0) atomicMin(&mm[12], fkey(sred[0]));
  __syncthreads();
  sred[tid] = lmx; __syncthreads();
  for (int st = 128; st > 0; st >>= 1) { if (tid < st) sred[tid] = fmaxf(sred[tid], sred[tid+st]); __syncthreads(); }
  if (tid == 0) atomicMax(&mm[13], fkey(sred[0]));
}

// ---------------------------------------------------------------- kernel 10: final norm01 in place
__global__ void __launch_bounds__(256)
k_final(float* ws, float* out) {
  unsigned* mm = (unsigned*)(ws + OFF_MM);
  float mn = funkey(mm[12]);
  float inv = 1.0f / (funkey(mm[13]) - mn);
  size_t i = (size_t)blockIdx.x * 256 + threadIdx.x;
  out[i] = (out[i] - mn) * inv;
}

// ---------------------------------------------------------------- launch
extern "C" void kernel_launch(void* const* d_in, const int* in_sizes, int n_in,
                              void* d_out, int out_size, void* d_ws, size_t ws_size,
                              hipStream_t stream) {
  const float* x      = (const float*)d_in[0];
  const float* latent = (const float*)d_in[1];
  const float* gw  = (const float*)d_in[2];  const float* gb  = (const float*)d_in[3];
  const float* wbw = (const float*)d_in[4];  const float* wbb = (const float*)d_in[5];
  const float* gaw = (const float*)d_in[6];  const float* gab = (const float*)d_in[7];
  const float* shw = (const float*)d_in[8];  const float* shb = (const float*)d_in[9];
  const float* dfw = (const float*)d_in[10]; const float* dfb = (const float*)d_in[11];
  const float* ctw = (const float*)d_in[12]; const float* ctb = (const float*)d_in[13];
  const float* tow = (const float*)d_in[14]; const float* tob = (const float*)d_in[15];
  const float* ffw = (const float*)d_in[16]; const float* ffb = (const float*)d_in[17];
  float* ws  = (float*)d_ws;
  float* out = (float*)d_out;

  k_init<<<1, 256, 0, stream>>>(ws);
  k_params<<<1, 32, 0, stream>>>(latent, gw, gb, wbw, wbb, gaw, gab, shw, shb,
                                 dfw, dfb, ctw, ctb, tow, tob, ffw, ffb, ws);
  k_blur<<<dim3(16, 16, NPLANE), 256, 0, stream>>>(x, ws);
  k_dark<<<dim3(HW/256, BB), 256, 0, stream>>>(x, ws);
  k_select<<<BB, 256, 0, stream>>>(ws);
  k_atmo<<<dim3(HW/256, BB), 256, 0, stream>>>(x, ws);
  k_fft_rows<<<NPLANE*HH, 256, 0, stream>>>(x, ws);
  k_fft_cols<<<NPLANE*WW, 256, 0, stream>>>(ws);
  k_stats<<<BB*HW/256, 256, 0, stream>>>(x, ws);
  k_sum<<<NPLANE*HH, 256, 0, stream>>>(x, ws, out);
  k_final<<<NTOT/256, 256, 0, stream>>>(ws, out);
}
